// GCN_80075370266806
// MI455X (gfx1250) — matrix operand order repaired and checked by reading, not compiled, not run on hardware
//
#include <hip/hip_runtime.h>

typedef __attribute__((ext_vector_type(16))) _Float16 v16h;
typedef __attribute__((ext_vector_type(8)))  float    v8f;
typedef __attribute__((ext_vector_type(4)))  float    f32x4;

#define NFEAT 128

// ---------------- utility kernels ----------------

__global__ void zero_f32(float* __restrict__ p, int n) {
    for (int i = blockIdx.x * blockDim.x + threadIdx.x; i < n; i += gridDim.x * blockDim.x)
        p[i] = 0.0f;
}

__global__ void init_out(float* __restrict__ out, const float* __restrict__ b2, int n) {
    for (int i = blockIdx.x * blockDim.x + threadIdx.x; i < n; i += gridDim.x * blockDim.x)
        out[i] = b2[i & (NFEAT - 1)];
}

__global__ void deg_count(const long long* __restrict__ eidx, float* __restrict__ deg, int E) {
    int e = blockIdx.x * blockDim.x + threadIdx.x;
    if (e < E) {
        int col = (int)eidx[E + e];   // target node
        atomicAdd(&deg[col], 1.0f);
    }
}

__global__ void finish_deg(float* __restrict__ deg, int n) {
    int i = blockIdx.x * blockDim.x + threadIdx.x;
    if (i < n) {
        float d = deg[i];
        deg[i] = (d > 0.0f) ? rsqrtf(d) : 0.0f;   // deg becomes deg_inv_sqrt in place
    }
}

// Pack W[K=128][N=128] (row-major) into per-lane B-fragment order:
// index = ((tileN*4 + kstep)*32 + lane)*16 + i ; element = W[kstep*32 + (lane>=16)*8 + (i&7) + (i>>3)*16][tileN*16 + (lane&15)]
__global__ void pack_w(const float* __restrict__ W, _Float16* __restrict__ wp) {
    int idx = blockIdx.x * blockDim.x + threadIdx.x;   // 0 .. 16383
    if (idx >= NFEAT * NFEAT) return;
    int i    = idx & 15;
    int lane = (idx >> 4) & 31;
    int ks   = (idx >> 9) & 3;
    int tn   = idx >> 11;
    int n = tn * 16 + (lane & 15);
    int k = ks * 32 + ((lane >> 4) << 3) + (i & 7) + ((i >> 3) << 4);
    wp[idx] = (_Float16)W[k * NFEAT + n];
}

// ---------------- WMMA GEMM: C[M,128] = f(A[M,128]) @ W[128,128] ----------------
// One wave32 per 16x16 output tile. K=128 as 4 steps of v_wmma_f32_16x16x32_f16.
// mode==1: A element transform relu(a + bias[k]) (layer-2 input epilogue fused on load).
__global__ void gemm_wmma(const float* __restrict__ A, const _Float16* __restrict__ Wp,
                          const float* __restrict__ bias, float* __restrict__ C,
                          int M, int mode) {
    int lane = threadIdx.x & 31;
    int warp = threadIdx.x >> 5;
    int wid  = blockIdx.x * (blockDim.x >> 5) + warp;
    int nTilesN = NFEAT / 16;                      // 8
    int totalWaves = (M / 16) * nTilesN;
    if (wid >= totalWaves) return;
    int mTile = wid >> 3;
    int nTile = wid & 7;

    int mrow  = mTile * 16 + (lane & 15);
    int kbase = (lane >> 4) << 3;                  // 0 or 8

    v8f acc = {};
    #pragma unroll
    for (int ks = 0; ks < 4; ++ks) {
        int k0 = ks * 32;
        const float* arow = A + (size_t)mrow * NFEAT + k0 + kbase;
        f32x4 p0 = *(const f32x4*)(arow);
        f32x4 p1 = *(const f32x4*)(arow + 4);
        f32x4 p2 = *(const f32x4*)(arow + 16);
        f32x4 p3 = *(const f32x4*)(arow + 20);
        float f[16] = { p0.x, p0.y, p0.z, p0.w, p1.x, p1.y, p1.z, p1.w,
                        p2.x, p2.y, p2.z, p2.w, p3.x, p3.y, p3.z, p3.w };
        v16h a;
        if (mode) {
            #pragma unroll
            for (int i = 0; i < 8; ++i) {
                int k = k0 + kbase + i;
                a[i]     = (_Float16)fmaxf(f[i]     + bias[k],      0.0f);
                a[8 + i] = (_Float16)fmaxf(f[8 + i] + bias[k + 16], 0.0f);
            }
        } else {
            #pragma unroll
            for (int i = 0; i < 16; ++i) a[i] = (_Float16)f[i];
        }
        // B fragment: 16 contiguous halfs per lane, prepacked
        v16h b = *(const v16h*)(Wp + (((size_t)nTile * 4 + ks) * 32 + lane) * 16);
        acc = __builtin_amdgcn_wmma_f32_16x16x32_f16(false, a, false, b,
                                                     (short)0, acc, false, false);
    }

    // C/D layout: lanes 0-15 -> rows 0..7 (vgpr r), lanes 16-31 -> rows 8..15; col = lane&15
    int col   = nTile * 16 + (lane & 15);
    int rbase = (lane >> 4) << 3;
    #pragma unroll
    for (int r = 0; r < 8; ++r)
        C[(size_t)(mTile * 16 + rbase + r) * NFEAT + col] = acc[r];
}

// ---------------- edge aggregation: out[col] += h[row] * dinv[row]*dinv[col] ----------------
// One wave32 per edge; each lane handles 4 consecutive features (float4 gather, 4 f32 atomics).
__global__ void aggregate(const long long* __restrict__ eidx, const float* __restrict__ dinv,
                          const float* __restrict__ h, float* __restrict__ out, int E) {
    int lane = threadIdx.x & 31;
    int warp = threadIdx.x >> 5;
    int wpb  = blockDim.x >> 5;
    for (int e = blockIdx.x * wpb + warp; e < E; e += gridDim.x * wpb) {
        int row = (int)eidx[e];
        int col = (int)eidx[E + e];
        float norm = dinv[row] * dinv[col];
        const float* src = h + (size_t)row * NFEAT + lane * 4;
        f32x4 v = *(const f32x4*)src;
        float* dst = out + (size_t)col * NFEAT + lane * 4;
        atomicAdd(dst + 0, v.x * norm);
        atomicAdd(dst + 1, v.y * norm);
        atomicAdd(dst + 2, v.z * norm);
        atomicAdd(dst + 3, v.w * norm);
    }
}

// ---------------- launch ----------------

extern "C" void kernel_launch(void* const* d_in, const int* in_sizes, int n_in,
                              void* d_out, int out_size, void* d_ws, size_t ws_size,
                              hipStream_t stream) {
    const float*     x    = (const float*)d_in[0];
    const long long* eidx = (const long long*)d_in[1];   // int64 edge_index [2,E]
    const float*     W1   = (const float*)d_in[2];
    const float*     b1   = (const float*)d_in[3];
    const float*     W2   = (const float*)d_in[4];
    const float*     b2   = (const float*)d_in[5];
    float*           out  = (float*)d_out;

    int N = in_sizes[0] / NFEAT;     // 50000
    int E = in_sizes[1] / 2;         // 625000

    char* ws = (char*)d_ws;
    size_t off = 0;
    float* dinv = (float*)(ws + off);            off += ((size_t)N * 4 + 255) & ~(size_t)255;
    _Float16* wp1 = (_Float16*)(ws + off);       off += NFEAT * NFEAT * sizeof(_Float16);
    _Float16* wp2 = (_Float16*)(ws + off);       off += NFEAT * NFEAT * sizeof(_Float16);
    float* hbuf   = (float*)(ws + off);          off += (size_t)N * NFEAT * 4;
    float* aggbuf = (float*)(ws + off);          off += (size_t)N * NFEAT * 4;

    const int BT = 256;                           // 8 wave32 per block
    int nElems = N * NFEAT;
    int gemmBlocks = ((N / 16) * (NFEAT / 16) + (BT / 32) - 1) / (BT / 32);

    // normalization coefficients
    zero_f32  <<<256, BT, 0, stream>>>(dinv, N);
    deg_count <<<(E + BT - 1) / BT, BT, 0, stream>>>(eidx, dinv, E);
    finish_deg<<<(N + BT - 1) / BT, BT, 0, stream>>>(dinv, N);

    // prepack both weight matrices into WMMA B-fragment order
    pack_w<<<(NFEAT * NFEAT + BT - 1) / BT, BT, 0, stream>>>(W1, wp1);
    pack_w<<<(NFEAT * NFEAT + BT - 1) / BT, BT, 0, stream>>>(W2, wp2);

    // layer 1: h = x @ W1 ; agg1 = scatter(norm * gather(h)) ; (bias+relu fused into layer-2 A load)
    gemm_wmma<<<gemmBlocks, BT, 0, stream>>>(x, wp1, nullptr, hbuf, N, 0);
    zero_f32 <<<2048, BT, 0, stream>>>(aggbuf, nElems);
    aggregate<<<2048, BT, 0, stream>>>(eidx, dinv, hbuf, aggbuf, E);

    // layer 2: h2 = relu(agg1 + b1) @ W2 ; out = b2 + scatter(norm * gather(h2))
    gemm_wmma<<<gemmBlocks, BT, 0, stream>>>(aggbuf, wp2, b1, hbuf, N, 1);
    init_out <<<2048, BT, 0, stream>>>(out, b2, nElems);
    aggregate<<<2048, BT, 0, stream>>>(eidx, dinv, hbuf, out, E);
}
